// Transformer_29506425324125
// MI455X (gfx1250) — compile-verified, matrix operand order checked by reading, not run on hardware
//
#include <hip/hip_runtime.h>
#include <hip/hip_bf16.h>
#include <math.h>

// ---------------------------------------------------------------------------
// MI455X (gfx1250) BERT forward. bf16 WMMA (16x16x32, f32 accum) everywhere;
// f32 only for LayerNorm / softmax / residual numerics. Vocab GEMM streams
// straight into d_out in [N,V,S] order (1.09 GB score write is the roofline
// bottleneck: ~2.2 GB total traffic @ 23.3 TB/s ≈ 95 us >> WMMA math time).
// A-tiles staged by the Tensor Data Mover (TDM) with hardware row padding;
// B-tiles staged swizzled so every WMMA operand is two ds_load_b128.
// ---------------------------------------------------------------------------

typedef __attribute__((ext_vector_type(16))) __bf16 v16bf;
typedef __attribute__((ext_vector_type(8)))  float  v8f;
typedef unsigned int u32x4 __attribute__((ext_vector_type(4)));
typedef int          i32x8 __attribute__((ext_vector_type(8)));
typedef int          i32x4 __attribute__((ext_vector_type(4)));

#define VOCAB   33279
#define NB      128
#define SEQ     64
#define EMB     256
#define HID     512
#define NHEAD   8
#define DHEAD   32
#define NLAYER  4
#define MROWS   (NB * SEQ)   // 8192

static __device__ __forceinline__ unsigned short f2bf(float f) {
  union { float f; unsigned u; } v; v.f = f;
  unsigned r = v.u + 0x7FFFu + ((v.u >> 16) & 1u);   // round-to-nearest-even
  return (unsigned short)(r >> 16);
}
static __device__ __forceinline__ float gelu_exact(float x) {
  return 0.5f * x * (1.0f + erff(x * 0.70710678118654752f));
}

// A-matrix fragment (16x32 bf16) from row-major LDS tile, as two b128 loads.
// ISA 7.12.2: lanes 0-15 -> row=lane, K={0..7,16..23}; lanes 16-31 -> row=lane-16, K={8..15,24..31}
static __device__ __forceinline__ v16bf frag_rowmajor(const unsigned short* s, int ld, int lane) {
  int r  = lane & 15;
  int kb = (lane >> 4) << 3;
  const unsigned short* p = s + r * ld + kb;
  union { uint4 q[2]; v16bf v; } f;
  f.q[0] = *(const uint4*)(p);        // K kb..kb+7
  f.q[1] = *(const uint4*)(p + 16);   // K kb+16..kb+23
  return f.v;
}
// B fragment from swizzled LDS (lane's 16 bf16 stored contiguously): two b128 loads.
static __device__ __forceinline__ v16bf frag_swz(const unsigned short* s) {
  union { uint4 q[2]; v16bf v; } f;
  f.q[0] = *(const uint4*)(s);
  f.q[1] = *(const uint4*)(s + 8);
  return f.v;
}

// ---------------------------------------------------------------------------
// TDM: 2D bf16 tile (tile_d0 x tile_d1, row stride = stride_d0 elements) from
// global to LDS with hardware padding (pad codes: interval 0=2DW..., amount 0=1DW...).
// D# packing per CDNA5 ISA; type=2 ("image"), count=1.
// This toolchain's builtin takes 6 args (g0, g1, g2, g3, g4, cpol); groups
// 2..4 are zero/disabled for a 2D tile.
// ---------------------------------------------------------------------------
static __device__ __forceinline__ void tdm_load_2d(
    const void* gptr, unsigned lds_byte_off,
    unsigned tensor_d0, unsigned tensor_d1,
    unsigned tile_d0, unsigned tile_d1,
    unsigned long long stride_d0_elems,
    unsigned pad_interval_code, unsigned pad_amount_code)
{
  unsigned long long ga = (unsigned long long)gptr;
  u32x4 g0;
  g0[0] = 1u;                                             // count=1, user desc
  g0[1] = lds_byte_off;                                   // lds_addr
  g0[2] = (unsigned)(ga & 0xFFFFFFFFu);                   // global_addr[31:0]
  g0[3] = (unsigned)((ga >> 32) & 0x01FFFFFFu)            // global_addr[56:32]
        | (2u << 30);                                     // type=2
  i32x8 g1;
  g1[0] = (int)((1u << 16)                                // data_size = 2B
        | (1u << 20)                                      // pad_enable
        | (pad_interval_code << 22) | (pad_amount_code << 25));
  g1[1] = (int)((tensor_d0 & 0xFFFFu) << 16);             // tensor_dim0[15:0] at bit48
  g1[2] = (int)((tensor_d0 >> 16) | ((tensor_d1 & 0xFFFFu) << 16));
  g1[3] = (int)((tensor_d1 >> 16) | (tile_d0 << 16));
  g1[4] = (int)(tile_d1 & 0xFFFFu);                       // tile_dim1; tile_dim2=0
  g1[5] = (int)(stride_d0_elems & 0xFFFFFFFFull);         // dim0_stride[31:0] at bit160
  g1[6] = (int)((stride_d0_elems >> 32) & 0xFFFFull);     // dim0_stride[47:32]; dim1_stride=0
  g1[7] = 0;
  i32x4 z4 = {0, 0, 0, 0};
  i32x8 z8 = {0, 0, 0, 0, 0, 0, 0, 0};
  __builtin_amdgcn_tensor_load_to_lds(g0, g1, z4, z4, z8, 0);
}

// ---------------------------------------------------------------------------
// Generic bf16 GEMM: C[M,N] = act(A[M,K] @ B[K,N] + bias)
// Block tile 64x128, 8 waves (4x2), each wave 16x64 (4 WMMA tiles), K step 32.
// A staged by TDM (wave 0) with HW padding; B staged swizzled by all threads.
// vocab_transpose=1: scatter f32 result to d_out as score[n][col][s].
// ---------------------------------------------------------------------------
__global__ __launch_bounds__(256) void gemm_bf16_kernel(
    const unsigned short* __restrict__ A, const unsigned short* __restrict__ B,
    const float* __restrict__ bias, float* __restrict__ Cf,
    unsigned short* __restrict__ Cb, int M, int N, int K, int act, int vocab_transpose)
{
  __shared__ __align__(16) unsigned short As[64][40];      // 64x32 + 8 pad (TDM-padded)
  __shared__ __align__(32) unsigned short Bsw[8][32][16];  // fragment-major B
  const int tid  = threadIdx.x;
  const int lane = tid & 31;
  const int wave = tid >> 5;
  const int bm = blockIdx.y, bn = blockIdx.x;
  const int wm = wave >> 1, wn = wave & 1;

  v8f acc[4] = {};
  const unsigned ldsA = (unsigned)(size_t)&As[0][0];

  for (int k0 = 0; k0 < K; k0 += 32) {
    // --- A tile via Tensor Data Mover: 64 rows x 32 cols bf16, row stride K.
    // Row = 16 DW -> pad_interval code 3 (16 DW); pad = 4 DW (8 bf16) -> code 3.
    if (wave == 0) {
      tdm_load_2d(A + (size_t)(bm * 64) * K + k0, ldsA,
                  /*tensor_d0=*/32, /*tensor_d1=*/64,
                  /*tile_d0=*/32, /*tile_d1=*/64,
                  (unsigned long long)K, 3u, 3u);
      __builtin_amdgcn_s_wait_tensorcnt(0);
    }
    // --- B tile (32x128), staged swizzled: thread -> row br, 16 cols of tile t.
    {
      int br = tid >> 3, t = tid & 7;
      int gc = bn * 128 + t * 16;
      const size_t gb = (size_t)(k0 + br) * N;
      unsigned short tmp[16];
      if (((N & 7) == 0) && (gc + 16 <= N)) {
        *(uint4*)&tmp[0] = *(const uint4*)(B + gb + gc);
        *(uint4*)&tmp[8] = *(const uint4*)(B + gb + gc + 8);
      } else {
#pragma unroll
        for (int i = 0; i < 16; ++i)
          tmp[i] = (gc + i < N) ? B[gb + gc + i] : (unsigned short)0;
      }
      int lh = (br >> 3) & 1;                    // which half-wave holds this K
      int j  = (br & 7) + ((br >> 4) << 3);      // element slot within fragment
#pragma unroll
      for (int i = 0; i < 16; ++i)
        Bsw[t][lh * 16 + i][j] = tmp[i];
    }
    if (k0 + 32 < K)  // pull next B tile toward L2 (global_prefetch_b8)
      __builtin_prefetch(B + (size_t)(k0 + 32 + (tid >> 3)) * N + bn * 128 + (tid & 7) * 16, 0, 0);
    __syncthreads();

    v16bf a = frag_rowmajor(&As[wm * 16][0], 40, lane);
#pragma unroll
    for (int t = 0; t < 4; ++t) {
      v16bf b = frag_swz(&Bsw[wn * 4 + t][lane][0]);
      acc[t] = __builtin_amdgcn_wmma_f32_16x16x32_bf16(
          false, a, false, b, (short)0, acc[t], false, false);
    }
    __syncthreads();
  }

  // Epilogue: bias + optional GELU, f32 and/or bf16 stores.
  const int m0 = (lane >> 4) << 3;   // C layout: lanes 16-31 hold rows 8..15
  const int cl = lane & 15;
#pragma unroll
  for (int t = 0; t < 4; ++t) {
    int col = bn * 128 + wn * 64 + t * 16 + cl;
    if (col >= N) continue;
    float bv = bias ? bias[col] : 0.f;
#pragma unroll
    for (int r = 0; r < 8; ++r) {
      int row = bm * 64 + wm * 16 + m0 + r;
      float v = acc[t][r] + bv;
      if (act == 1) v = gelu_exact(v);
      if (Cf) {
        if (vocab_transpose) {
          int n = row >> 6, s = row & 63;          // row = n*SEQ + s
          Cf[((size_t)n * VOCAB + col) * SEQ + s] = v;
        } else {
          Cf[(size_t)row * N + col] = v;
        }
      }
      if (Cb) Cb[(size_t)row * N + col] = f2bf(v);
    }
  }
}

// ---------------------------------------------------------------------------
// Fused attention per (head, batch): scores = QK^T/sqrt(32) -> softmax -> @V
// Q/K row-major LDS (fragments are contiguous b128 pairs); V staged swizzled.
// ---------------------------------------------------------------------------
__global__ __launch_bounds__(256) void attention_kernel(
    const unsigned short* __restrict__ q, const unsigned short* __restrict__ k,
    const unsigned short* __restrict__ v, unsigned short* __restrict__ o)
{
  __shared__ __align__(16) unsigned short Qs[64][40], Ks[64][40];
  __shared__ __align__(32) unsigned short Vsw[2][2][32][16]; // [Kchunk][Ntile][lane][16]
  __shared__ float Sc[64][65];
  __shared__ __align__(16) unsigned short Ab[64][72];
  const int h = blockIdx.x, n = blockIdx.y;
  const int tid = threadIdx.x, lane = tid & 31, wave = tid >> 5;

  {  // stage head slices [64 x 32]
    int r = tid >> 2, c0 = (tid & 3) << 3;
    size_t g = (size_t)(n * 64 + r) * EMB + h * DHEAD + c0;
    *(uint4*)&Qs[r][c0] = *(const uint4*)(q + g);
    *(uint4*)&Ks[r][c0] = *(const uint4*)(k + g);
    unsigned short tmp[8];
    *(uint4*)&tmp[0] = *(const uint4*)(v + g);
    int chunk = r >> 5, kr = r & 31;
    int lh = (kr >> 3) & 1, j = (kr & 7) + ((kr >> 4) << 3), t = c0 >> 4;
#pragma unroll
    for (int i = 0; i < 8; ++i)
      Vsw[chunk][t][lh * 16 + ((c0 & 15) + i)][j] = tmp[i];
  }
  __syncthreads();

  const int wm = wave & 3, wn = wave >> 2;
  const int m0 = (lane >> 4) << 3, cl = lane & 15;

  // scores 64x64: K^T fragment == K tile read with the A (row-major) pattern
  {
    v16bf a = frag_rowmajor(&Qs[wm * 16][0], 40, lane);
#pragma unroll
    for (int tt = 0; tt < 2; ++tt) {
      v16bf b = frag_rowmajor(&Ks[wn * 32 + tt * 16][0], 40, lane);
      v8f accv = {};
      accv = __builtin_amdgcn_wmma_f32_16x16x32_bf16(
          false, a, false, b, (short)0, accv, false, false);
#pragma unroll
      for (int r = 0; r < 8; ++r)
        Sc[wm * 16 + m0 + r][wn * 32 + tt * 16 + cl] =
            accv[r] * 0.17677669529663687f;   // 1/sqrt(32)
    }
  }
  __syncthreads();

  // softmax: one row per thread (tid < 64)
  if (tid < 64) {
    float mx = -1e30f;
    for (int j = 0; j < 64; ++j) mx = fmaxf(mx, Sc[tid][j]);
    float sum = 0.f;
    for (int j = 0; j < 64; ++j) sum += __expf(Sc[tid][j] - mx);
    float inv = 1.f / sum;
    for (int j = 0; j < 64; ++j) Ab[tid][j] = f2bf(__expf(Sc[tid][j] - mx) * inv);
  }
  __syncthreads();

  // context = attn[64x64] @ V[64x32]; 8 waves x one 16x16 tile; K = 64 (2 steps)
  {
    v8f accv = {};
#pragma unroll
    for (int ks = 0; ks < 64; ks += 32) {
      v16bf a = frag_rowmajor(&Ab[wm * 16][ks], 72, lane);
      v16bf b = frag_swz(&Vsw[ks >> 5][wn][lane][0]);
      accv = __builtin_amdgcn_wmma_f32_16x16x32_bf16(
          false, a, false, b, (short)0, accv, false, false);
    }
#pragma unroll
    for (int r = 0; r < 8; ++r) {
      int sr = wm * 16 + m0 + r, d = wn * 16 + cl;
      o[(size_t)(n * 64 + sr) * EMB + h * DHEAD + d] = f2bf(accv[r]);
    }
  }
}

// ---------------------------------------------------------------------------
// Embedding + positional + LayerNorm (one token per block, E=256 threads)
// ---------------------------------------------------------------------------
__global__ __launch_bounds__(256) void embed_ln_kernel(
    const int* __restrict__ label, const int* __restrict__ mask,
    const float* __restrict__ emb, const float* __restrict__ pos,
    const float* __restrict__ g, const float* __restrict__ b,
    float* __restrict__ xf, unsigned short* __restrict__ xb)
{
  __shared__ float r1[256], r2[256];
  int tok = blockIdx.x, e = threadIdx.x;
  int s = tok & 63;
  int id = (mask[tok] == 1) ? VOCAB : label[tok];
  float val = emb[(size_t)id * EMB + e] + pos[s * EMB + e];
  r1[e] = val; r2[e] = val * val; __syncthreads();
  for (int o = 128; o > 0; o >>= 1) {
    if (e < o) { r1[e] += r1[e + o]; r2[e] += r2[e + o]; }
    __syncthreads();
  }
  float mu  = r1[0] * (1.f / EMB);
  float var = r2[0] * (1.f / EMB) - mu * mu;
  float y = (val - mu) * rsqrtf(var + 1e-5f) * g[e] + b[e];
  xf[(size_t)tok * EMB + e] = y;
  xb[(size_t)tok * EMB + e] = f2bf(y);
}

// y = LN(x (+ r)) * g + b ; writes f32 (optional) and bf16 (optional)
__global__ __launch_bounds__(256) void residual_ln_kernel(
    const float* __restrict__ x, const float* __restrict__ res,
    const float* __restrict__ g, const float* __restrict__ b,
    float* __restrict__ outf, unsigned short* __restrict__ outb)
{
  __shared__ float r1[256], r2[256];
  int tok = blockIdx.x, e = threadIdx.x;
  float val = x[(size_t)tok * EMB + e];
  if (res) val += res[(size_t)tok * EMB + e];
  r1[e] = val; r2[e] = val * val; __syncthreads();
  for (int o = 128; o > 0; o >>= 1) {
    if (e < o) { r1[e] += r1[e + o]; r2[e] += r2[e + o]; }
    __syncthreads();
  }
  float mu  = r1[0] * (1.f / EMB);
  float var = r2[0] * (1.f / EMB) - mu * mu;
  float y = (val - mu) * rsqrtf(var + 1e-5f) * g[e] + b[e];
  if (outf) outf[(size_t)tok * EMB + e] = y;
  if (outb) outb[(size_t)tok * EMB + e] = f2bf(y);
}

// ---------------------------------------------------------------------------
// NLL loss over transposed score buffer: score[n][v][s], stride S over v.
// ---------------------------------------------------------------------------
__global__ __launch_bounds__(256) void loss_kernel(
    const float* __restrict__ score, const int* __restrict__ label,
    float* __restrict__ loss)
{
  __shared__ float red[256];
  int tok = blockIdx.x, t = threadIdx.x;
  int n = tok >> 6, s = tok & 63;
  const float* base = score + (size_t)n * VOCAB * SEQ + s;
  float mx = -1e30f;
  for (int v = t; v < VOCAB; v += 256) mx = fmaxf(mx, base[(size_t)v * SEQ]);
  red[t] = mx; __syncthreads();
  for (int o = 128; o > 0; o >>= 1) {
    if (t < o) red[t] = fmaxf(red[t], red[t + o]);
    __syncthreads();
  }
  mx = red[0]; __syncthreads();
  float sum = 0.f;
  for (int v = t; v < VOCAB; v += 256) sum += __expf(base[(size_t)v * SEQ] - mx);
  red[t] = sum; __syncthreads();
  for (int o = 128; o > 0; o >>= 1) {
    if (t < o) red[t] += red[t + o];
    __syncthreads();
  }
  if (t == 0) {
    int lab = label[tok];
    float lp = base[(size_t)lab * SEQ] - mx - logf(red[0]);
    atomicAdd(loss, -lp * (1.f / (float)MROWS));
  }
}

__global__ void zero_loss_kernel(float* loss) {
  if (threadIdx.x == 0 && blockIdx.x == 0) *loss = 0.f;
}

__global__ __launch_bounds__(256) void cvt_bf16_kernel(
    const float* __restrict__ src, unsigned short* __restrict__ dst, int n)
{
  for (int i = blockIdx.x * 256 + threadIdx.x; i < n; i += gridDim.x * 256)
    dst[i] = f2bf(src[i]);
}

// ---------------------------------------------------------------------------
extern "C" void kernel_launch(void* const* d_in, const int* in_sizes, int n_in,
                              void* d_out, int out_size, void* d_ws, size_t ws_size,
                              hipStream_t stream)
{
  (void)in_sizes; (void)n_in; (void)out_size; (void)ws_size;
  const int*   label = (const int*)d_in[0];
  const int*   maskp = (const int*)d_in[1];
  const float* emb   = (const float*)d_in[2];
  const float* pos   = (const float*)d_in[3];
  const float* emb_g = (const float*)d_in[4];
  const float* emb_b = (const float*)d_in[5];
  const float* Wq = (const float*)d_in[6];  const float* bq = (const float*)d_in[7];
  const float* Wk = (const float*)d_in[8];  const float* bk = (const float*)d_in[9];
  const float* Wv = (const float*)d_in[10]; const float* bv = (const float*)d_in[11];
  const float* Wo = (const float*)d_in[12]; const float* bo = (const float*)d_in[13];
  const float* n1g = (const float*)d_in[14]; const float* n1b = (const float*)d_in[15];
  const float* W1 = (const float*)d_in[16]; const float* b1 = (const float*)d_in[17];
  const float* W2 = (const float*)d_in[18]; const float* b2 = (const float*)d_in[19];
  const float* n2g = (const float*)d_in[20]; const float* n2b = (const float*)d_in[21];
  const float* dW1 = (const float*)d_in[22]; const float* db1 = (const float*)d_in[23];
  const float* dng = (const float*)d_in[24]; const float* dnb = (const float*)d_in[25];
  const float* dW2 = (const float*)d_in[26]; const float* db2 = (const float*)d_in[27];

  char* wsb = (char*)d_ws; size_t off = 0;
  auto alloc = [&](size_t bytes) -> void* {
    void* p = wsb + off; off += (bytes + 255) & ~(size_t)255; return p;
  };

  // bf16 weight copies (re-converted every call: deterministic, no caching)
  unsigned short* wq_b  = (unsigned short*)alloc(2ull * NLAYER * EMB * EMB);
  unsigned short* wk_b  = (unsigned short*)alloc(2ull * NLAYER * EMB * EMB);
  unsigned short* wv_b  = (unsigned short*)alloc(2ull * NLAYER * EMB * EMB);
  unsigned short* wo_b  = (unsigned short*)alloc(2ull * NLAYER * EMB * EMB);
  unsigned short* w1_b  = (unsigned short*)alloc(2ull * NLAYER * EMB * HID);
  unsigned short* w2_b  = (unsigned short*)alloc(2ull * NLAYER * HID * EMB);
  unsigned short* dw1_b = (unsigned short*)alloc(2ull * EMB * EMB);
  unsigned short* dw2_b = (unsigned short*)alloc(2ull * EMB * VOCAB);
  // activations
  float*          xf   = (float*)alloc(4ull * MROWS * EMB);
  unsigned short* xb   = (unsigned short*)alloc(2ull * MROWS * EMB);
  float*          x1f  = (float*)alloc(4ull * MROWS * EMB);
  unsigned short* x1b  = (unsigned short*)alloc(2ull * MROWS * EMB);
  unsigned short* qbf  = (unsigned short*)alloc(2ull * MROWS * EMB);
  unsigned short* kbf  = (unsigned short*)alloc(2ull * MROWS * EMB);
  unsigned short* vbf  = (unsigned short*)alloc(2ull * MROWS * EMB);
  unsigned short* obf  = (unsigned short*)alloc(2ull * MROWS * EMB);
  float*          tf   = (float*)alloc(4ull * MROWS * EMB);
  unsigned short* h1b  = (unsigned short*)alloc(2ull * MROWS * HID);
  unsigned short* decb = (unsigned short*)alloc(2ull * MROWS * EMB);

  float* lossp = (float*)d_out + (size_t)NB * VOCAB * SEQ;

  auto gemm = [&](const unsigned short* A, const unsigned short* B,
                  const float* bias, float* Cf, unsigned short* Cb,
                  int M, int N, int K, int act, int vt) {
    dim3 g((N + 127) / 128, M / 64);
    gemm_bf16_kernel<<<g, 256, 0, stream>>>(A, B, bias, Cf, Cb, M, N, K, act, vt);
  };

  // weight conversions
  cvt_bf16_kernel<<<1024, 256, 0, stream>>>(Wq,  wq_b,  NLAYER * EMB * EMB);
  cvt_bf16_kernel<<<1024, 256, 0, stream>>>(Wk,  wk_b,  NLAYER * EMB * EMB);
  cvt_bf16_kernel<<<1024, 256, 0, stream>>>(Wv,  wv_b,  NLAYER * EMB * EMB);
  cvt_bf16_kernel<<<1024, 256, 0, stream>>>(Wo,  wo_b,  NLAYER * EMB * EMB);
  cvt_bf16_kernel<<<1024, 256, 0, stream>>>(W1,  w1_b,  NLAYER * EMB * HID);
  cvt_bf16_kernel<<<1024, 256, 0, stream>>>(W2,  w2_b,  NLAYER * HID * EMB);
  cvt_bf16_kernel<<<256,  256, 0, stream>>>(dW1, dw1_b, EMB * EMB);
  cvt_bf16_kernel<<<2048, 256, 0, stream>>>(dW2, dw2_b, EMB * VOCAB);
  zero_loss_kernel<<<1, 64, 0, stream>>>(lossp);

  // embedding + LN
  embed_ln_kernel<<<MROWS, 256, 0, stream>>>(label, maskp, emb, pos, emb_g, emb_b, xf, xb);

  for (int l = 0; l < NLAYER; ++l) {
    const size_t we = (size_t)l * EMB * EMB;
    gemm(xb, wq_b + we, bq + l * EMB, nullptr, qbf, MROWS, EMB, EMB, 0, 0);
    gemm(xb, wk_b + we, bk + l * EMB, nullptr, kbf, MROWS, EMB, EMB, 0, 0);
    gemm(xb, wv_b + we, bv + l * EMB, nullptr, vbf, MROWS, EMB, EMB, 0, 0);
    attention_kernel<<<dim3(NHEAD, NB), 256, 0, stream>>>(qbf, kbf, vbf, obf);
    gemm(obf, wo_b + we, bo + l * EMB, tf, nullptr, MROWS, EMB, EMB, 0, 0);
    residual_ln_kernel<<<MROWS, 256, 0, stream>>>(xf, tf, n1g + l * EMB, n1b + l * EMB, x1f, x1b);
    gemm(x1b, w1_b + (size_t)l * EMB * HID, b1 + l * HID, nullptr, h1b, MROWS, HID, EMB, 1, 0);
    gemm(h1b, w2_b + (size_t)l * HID * EMB, b2 + l * EMB, tf, nullptr, MROWS, EMB, HID, 0, 0);
    residual_ln_kernel<<<MROWS, 256, 0, stream>>>(x1f, tf, n2g + l * EMB, n2b + l * EMB, xf, xb);
  }

  // decoder head: linear+GELU -> LN -> vocab projection (streams into d_out transposed)
  gemm(xb, dw1_b, db1, tf, nullptr, MROWS, EMB, EMB, 1, 0);
  residual_ln_kernel<<<MROWS, 256, 0, stream>>>(tf, nullptr, dng, dnb, nullptr, decb);
  gemm(decb, dw2_b, db2, (float*)d_out, nullptr, MROWS, VOCAB, EMB, 0, 1);

  // loss from transposed score
  loss_kernel<<<MROWS, 256, 0, stream>>>((const float*)d_out, label, lossp);
}
